// MiniSTU_20787641712995
// MI455X (gfx1250) — matrix operand order repaired and checked by reading, not compiled, not run on hardware
//
#include <hip/hip_runtime.h>
#include <cstdint>

// ---------------------------------------------------------------------------
// MiniSTU for MI455X (gfx1250, wave32, WMMA + TDM + async-to-LDS).
//
//   Stage P (prep):  x -> bf16; Mcat^T[c][o][i] bf16 (c = 16 plus + 16 minus);
//                    Toeplitz A-tiles Wt[d_idx][c][16][32] bf16 (zero-padded
//                    outside the causal triangle; d = 16*(d_idx-1) = l0 - s0).
//   Stage G (gemm):  ZT[b][c][o][s] = bf16( x[b] @ Mcat_c ), f32 accumulation,
//                    v_wmma_f32_16x16x32_bf16; A tile staged by the Tensor
//                    Data Mover (tensor_load_to_lds, TENSORcnt, HW LDS pad),
//                    B tile by global_load_async_to_lds_b128 (ASYNCcnt);
//                    double-buffered.
//   Stage C (conv):  out[b][l][o] = sum_c sum_{s<=l} W_c[l-s] * ZT[b][c][o][s]
//                    as triangular blocked WMMA; b128 fragment loads from L2.
// ---------------------------------------------------------------------------

typedef __attribute__((ext_vector_type(16))) __bf16   v16bf;
typedef __attribute__((ext_vector_type(8)))  float    v8f;
typedef __attribute__((ext_vector_type(4)))  uint32_t v4u;
typedef __attribute__((ext_vector_type(8)))  uint32_t v8u;

#define SEQ_LEN 2048
#define C_TOT   32          // 16 plus-channels + 16 minus-channels
#define D_IN    512
#define D_OUT   512
#define NDIDX   129         // Toeplitz tile diagonals: d = -16, 0, 16, ..., 2032
#define GSTRIDE 40          // 80B LDS rows: 64B data + 16B pad (= TDM pad 16DW/4DW)

union Frag16 { uint32_t u[8]; uint4 q[2]; v16bf v; };

__device__ __forceinline__ uint16_t f2bf(float f) {
  uint32_t x = __float_as_uint(f);
  uint32_t r = (x + 0x7FFFu + ((x >> 16) & 1u)) >> 16;   // round-nearest-even
  return (uint16_t)r;
}

// CDNA5 async direct-to-LDS copy (16B per lane), tracked by ASYNCcnt.
__device__ __forceinline__ void async_ld_b128(uint32_t lds_off, const void* gaddr) {
  asm volatile("global_load_async_to_lds_b128 %0, %1, off"
               :: "v"(lds_off), "v"(gaddr) : "memory");
}
// CDNA5 Tensor Data Mover: one 2D-tile DMA per instruction, tracked by TENSORcnt.
__device__ __forceinline__ void tdm_load(v4u g0, v8u g1, v4u g2, v4u g3) {
  asm volatile("tensor_load_to_lds %0, %1, %2, %3"
               :: "s"(g0), "s"(g1), "s"(g2), "s"(g3) : "memory");
}
__device__ __forceinline__ uint32_t lds_off32(const void* p) {
  return (uint32_t)(uintptr_t)p;    // generic LDS ptr: low 32 bits = LDS offset
}

// ---------------- prep kernels ----------------

__global__ void prep_xb(const float* __restrict__ x, uint16_t* __restrict__ xb, int n) {
  int i = blockIdx.x * 256 + threadIdx.x;
  if (i < n) xb[i] = f2bf(x[i]);
}

// Mcat^T[c][o][i] = (c<16 ? Mp[c] : Mm[c-16])[i][o]
__global__ void prep_mcat(const float* __restrict__ Mp, const float* __restrict__ Mm,
                          uint16_t* __restrict__ mct, int n) {
  int i = blockIdx.x * 256 + threadIdx.x;
  if (i >= n) return;
  int o  = i & 511;
  int ii = (i >> 9) & 511;
  int c  = i >> 18;
  const float* M = (c < 16) ? Mp : Mm;
  int k = c & 15;
  float v = M[((size_t)k * D_IN + ii) * D_OUT + o];
  mct[((size_t)c * D_OUT + o) * D_IN + ii] = f2bf(v);
}

// Wt[d_idx][c][l(16)][ss(32)] = W_c[16*(d_idx-1) + l - ss], 0 outside [0,2048)
__global__ void prep_wtile(const float* __restrict__ phi, uint16_t* __restrict__ wt, int n) {
  int idx = blockIdx.x * 256 + threadIdx.x;
  if (idx >= n) return;
  int ss    = idx & 31;
  int l     = (idx >> 5) & 15;
  int c     = (idx >> 9) & 31;
  int d_idx = idx >> 14;
  int t = (d_idx - 1) * 16 + l - ss;
  float v = 0.0f;
  if (t >= 0 && t < SEQ_LEN) {
    v = phi[t * 16 + (c & 15)];
    if (c >= 16 && (t & 1)) v = -v;     // minus channel: (-1)^t * phi
  }
  wt[idx] = f2bf(v);
}

// ---------------- stage G: Z^T = (x @ Mcat)^T in bf16 ----------------

__global__ __launch_bounds__(256)
void gemm_z(const uint16_t* __restrict__ xb, const uint16_t* __restrict__ mct,
            uint16_t* __restrict__ zt) {
  __shared__ uint16_t Alds[2 * 128 * GSTRIDE];   // x tile   [l][i], double buffer
  __shared__ uint16_t Blds[2 * 128 * GSTRIDE];   // Mcat^T   [o][i], double buffer
  const int tid  = threadIdx.x;
  const int lane = tid & 31;
  const int half = lane >> 4;
  const int mn   = lane & 15;
  const int wave = tid >> 5;
  const int wr   = wave >> 2;                   // 0..1 : 64-row strip
  const int wc   = wave & 3;                    // 0..3 : 32-col strip
  const int b    = (int)blockIdx.z >> 5;
  const int c    = (int)blockIdx.z & 31;
  const int l0   = blockIdx.x * 128;
  const int o0   = blockIdx.y * 128;

  const uint16_t* Ag = xb  + ((size_t)b * SEQ_LEN + l0) * D_IN;
  const uint16_t* Bg = mct + ((size_t)c * D_OUT  + o0) * D_IN;

  // per-thread chunk: 2x (row, 8-col) pieces of the 128x32 B tile
  const int row0 = tid >> 2, col0 = (tid & 3) << 3;
  const int row1 = (tid + 256) >> 2, col1 = ((tid + 256) & 3) << 3;

  // TDM descriptor group 1 (constant): 2B elems, pad 4 DW after every 16 DW,
  // tensor 32 x 128 (row stride 512), tile 32 x 128.
  const v8u g1c = { (1u << 16) | (1u << 20) | (3u << 22) | (3u << 25),
                    32u << 16, 128u << 16, 32u << 16, 128u, 512u, 0u, 0u };
  const v4u gz = { 0u, 0u, 0u, 0u };

  auto issueA = [&](int buf, int kb) {          // whole A tile: one TDM op
    uint32_t lds_base = lds_off32(&Alds[buf * (128 * GSTRIDE)]);
    uint64_t ga = (uint64_t)(uintptr_t)(Ag + kb);
    v4u g0 = { 1u, lds_base, (uint32_t)ga,
               (uint32_t)((ga >> 32) & 0x01FFFFFFu) | (2u << 30) };
    tdm_load(g0, g1c, gz, gz);
  };
  auto issueB = [&](int buf, int kb) {          // B tile: per-lane async b128
    uint16_t* Bs = Blds + buf * (128 * GSTRIDE);
    async_ld_b128(lds_off32(&Bs[row0 * GSTRIDE + col0]), Bg + (size_t)row0 * D_IN + kb + col0);
    async_ld_b128(lds_off32(&Bs[row1 * GSTRIDE + col1]), Bg + (size_t)row1 * D_IN + kb + col1);
  };

  v8f acc[4][2] = {};

  issueB(0, 0);                                  // prologue: fill buffer 0
  if (wave == 0) issueA(0, 0);
  for (int step = 0; step < D_IN / 32; ++step) {
    const int kb = step * 32;
    if (step + 1 < D_IN / 32) {                  // prefetch next buffer
      issueB((step + 1) & 1, kb + 32);
      if (wave == 0) {
        issueA((step + 1) & 1, kb + 32);
        __builtin_amdgcn_s_wait_tensorcnt(1);    // current A tile landed
      }
      asm volatile("s_wait_asynccnt 0x2" ::: "memory");   // current B tile landed
    } else {
      if (wave == 0) __builtin_amdgcn_s_wait_tensorcnt(0);
      asm volatile("s_wait_asynccnt 0x0" ::: "memory");
    }
    __syncthreads();                             // publish all LDS writes

    const uint16_t* As = Alds + (step & 1) * (128 * GSTRIDE);
    const uint16_t* Bs = Blds + (step & 1) * (128 * GSTRIDE);
    Frag16 af[4], bfr[2];
#pragma unroll
    for (int mt = 0; mt < 4; ++mt) {
      const uint16_t* ar = &As[(wr * 64 + mt * 16 + mn) * GSTRIDE + half * 8];
      af[mt].q[0] = *(const uint4*)ar;           // K = half*8  + 0..7   (+0)
      af[mt].q[1] = *(const uint4*)(ar + 16);    // K = half*8  + 16..23 (+16)
    }
#pragma unroll
    for (int nt = 0; nt < 2; ++nt) {
      const uint16_t* br = &Bs[(wc * 32 + nt * 16 + mn) * GSTRIDE + half * 8];
      bfr[nt].q[0] = *(const uint4*)br;          // K = half*8 + 0..7
      bfr[nt].q[1] = *(const uint4*)(br + 16);   // K = half*8 + 16..23
    }
#pragma unroll
    for (int mt = 0; mt < 4; ++mt)
#pragma unroll
      for (int nt = 0; nt < 2; ++nt)
        acc[mt][nt] = __builtin_amdgcn_wmma_f32_16x16x32_bf16(
            false, af[mt].v, false, bfr[nt].v, (short)0, acc[mt][nt], false, false);

    __syncthreads();                             // protect buffer before rewrite
  }

  // D frag: lane (half,mn) holds M = half*8 + r, N = mn  ->  ZT row o, 8
  // consecutive s per lane -> one 16B store per tile.
  const size_t zrow = (size_t)(b * C_TOT + c) * D_OUT;
#pragma unroll
  for (int mt = 0; mt < 4; ++mt)
#pragma unroll
    for (int nt = 0; nt < 2; ++nt) {
      int o = o0 + wc * 32 + nt * 16 + mn;
      int s = l0 + wr * 64 + mt * 16 + half * 8;
      uint32_t p[4];
#pragma unroll
      for (int j = 0; j < 4; ++j)
        p[j] = (uint32_t)f2bf(acc[mt][nt][2 * j]) |
               ((uint32_t)f2bf(acc[mt][nt][2 * j + 1]) << 16);
      *(uint4*)(zt + (zrow + o) * SEQ_LEN + s) = make_uint4(p[0], p[1], p[2], p[3]);
    }
}

// ---------------- stage C: triangular Toeplitz conv ----------------

__global__ __launch_bounds__(256)
void conv_out_k(const uint16_t* __restrict__ wt, const uint16_t* __restrict__ zt,
                float* __restrict__ out) {
  const int tid  = threadIdx.x;
  const int lane = tid & 31;
  const int half = lane >> 4;
  const int mn   = lane & 15;
  const int wave = tid >> 5;
  const int wr   = wave >> 2;
  const int wc   = wave & 3;
  const int b    = blockIdx.z;
  const int l0   = blockIdx.x * 128;
  const int o0   = blockIdx.y * 128;

  v8f acc[4][2] = {};
  const int lw = l0 + wr * 64;
  const int sbCount = (lw + 48) / 32 + 1;        // cover s0 <= lw+48 (mt=3 diag)
  const size_t zbase = ((size_t)b * C_TOT * D_OUT + o0 + wc * 32) * SEQ_LEN;

  for (int sb = 0; sb < sbCount; ++sb) {
    const int s0 = sb * 32;
    if (sb + 1 < sbCount)                        // global_prefetch_b8 next strip
      __builtin_prefetch(zt + zbase + (size_t)mn * SEQ_LEN + s0 + 32, 0, 1);
    for (int c = 0; c < C_TOT; ++c) {
      const uint16_t* zp = zt + zbase + (size_t)c * ((size_t)D_OUT * SEQ_LEN);
      Frag16 bfr[2];
#pragma unroll
      for (int nt = 0; nt < 2; ++nt) {
        const uint16_t* zr = zp + (size_t)(nt * 16 + mn) * SEQ_LEN + s0 + half * 8;
        bfr[nt].q[0] = *(const uint4*)zr;        // K = half*8 + 0..7
        bfr[nt].q[1] = *(const uint4*)(zr + 16); // K = half*8 + 16..23
      }
#pragma unroll
      for (int mt = 0; mt < 4; ++mt) {
        const int d_idx = ((lw + mt * 16 - s0) >> 4) + 1;   // wave-uniform
        if (d_idx < 1) continue;                            // all-zero tile
        const uint16_t* wp = wt + (((size_t)d_idx * C_TOT + c) * 16 + mn) * 32 + half * 8;
        Frag16 af;
        af.q[0] = *(const uint4*)wp;             // K = half*8 + 0..7
        af.q[1] = *(const uint4*)(wp + 16);      // K = half*8 + 16..23
#pragma unroll
        for (int nt = 0; nt < 2; ++nt)
          acc[mt][nt] = __builtin_amdgcn_wmma_f32_16x16x32_bf16(
              false, af.v, false, bfr[nt].v, (short)0, acc[mt][nt], false, false);
      }
    }
  }

#pragma unroll
  for (int mt = 0; mt < 4; ++mt)
#pragma unroll
    for (int nt = 0; nt < 2; ++nt) {
      float* op = out + ((size_t)b * SEQ_LEN + lw + mt * 16 + half * 8) * D_OUT
                      + o0 + wc * 32 + nt * 16 + mn;
#pragma unroll
      for (int r = 0; r < 8; ++r) op[(size_t)r * D_OUT] = acc[mt][nt][r];
    }
}

// ---------------- launch ----------------

extern "C" void kernel_launch(void* const* d_in, const int* in_sizes, int n_in,
                              void* d_out, int out_size, void* d_ws, size_t ws_size,
                              hipStream_t stream) {
  const float* x   = (const float*)d_in[0];   // (2, 2048, 512)
  const float* phi = (const float*)d_in[1];   // (2048, 16)
  const float* Mp  = (const float*)d_in[2];   // (16, 512, 512)
  const float* Mm  = (const float*)d_in[3];   // (16, 512, 512)
  float* out = (float*)d_out;                 // (2, 2048, 512)

  const size_t sz_xb  = (size_t)2 * SEQ_LEN * D_IN * 2;           //   4 MiB
  const size_t sz_mct = (size_t)C_TOT * D_OUT * D_IN * 2;         //  16 MiB
  const size_t sz_wt  = (size_t)NDIDX * C_TOT * 16 * 32 * 2;      //  ~4 MiB
  const size_t off_xb  = 0;
  const size_t off_mct = off_xb + sz_xb;
  const size_t off_wt  = off_mct + sz_mct;
  const size_t off_zt  = (off_wt + sz_wt + 255) & ~(size_t)255;
  const size_t sz_zt   = (size_t)2 * C_TOT * D_OUT * SEQ_LEN * 2; // 128 MiB
  if (ws_size < off_zt + sz_zt) return;       // workspace too small

  uint16_t* xb  = (uint16_t*)((char*)d_ws + off_xb);
  uint16_t* mct = (uint16_t*)((char*)d_ws + off_mct);
  uint16_t* wt  = (uint16_t*)((char*)d_ws + off_wt);
  uint16_t* zt  = (uint16_t*)((char*)d_ws + off_zt);

  { int n = 2 * SEQ_LEN * D_IN;      prep_xb   <<<(n + 255) / 256, 256, 0, stream>>>(x, xb, n); }
  { int n = C_TOT * D_OUT * D_IN;    prep_mcat <<<(n + 255) / 256, 256, 0, stream>>>(Mp, Mm, mct, n); }
  { int n = NDIDX * C_TOT * 16 * 32; prep_wtile<<<(n + 255) / 256, 256, 0, stream>>>(phi, wt, n); }

  gemm_z    <<<dim3(SEQ_LEN / 128, D_OUT / 128, 2 * C_TOT), 256, 0, stream>>>(xb, mct, zt);
  conv_out_k<<<dim3(SEQ_LEN / 128, D_OUT / 128, 2),         256, 0, stream>>>(wt, zt, out);
}
